// DecoderRNN_LSTMCell_74964359184716
// MI455X (gfx1250) — matrix operand order repaired and checked by reading, not compiled, not run on hardware
//
#include <hip/hip_runtime.h>
#include <hip/hip_bf16.h>
#include <math.h>

// Problem constants (from reference): B=64, E=512, H=512, V=32000, T=32
#define BB 64
#define EE 512
#define HH 512
#define VV 32000
#define TT 32
#define G4H 2048   /* 4*H */
#define KC  1024   /* E+H concat K for gates GEMM */

typedef __attribute__((ext_vector_type(16))) __bf16 v16bf;
typedef __attribute__((ext_vector_type(8)))  __bf16 v8bf;
typedef __attribute__((ext_vector_type(8)))  float  v8f;

__device__ __forceinline__ __bf16 f2bf(float f) {
    unsigned int u; __builtin_memcpy(&u, &f, 4);
    u += 0x7FFFu + ((u >> 16) & 1u);           // round-to-nearest-even
    unsigned short s = (unsigned short)(u >> 16);
    __bf16 b; __builtin_memcpy(&b, &s, 2);
    return b;
}

__device__ __forceinline__ float sigm(float x) { return 1.0f / (1.0f + expf(-x)); }

// A-fragment (16x32 bf16 tile of row-major A, lda=K): lane holds row (m0+ml),
// K-chunks {16g..16g+15} (contiguous, matching B fragment).
__device__ __forceinline__ v16bf load_a_frag(const __bf16* __restrict__ A, int K,
                                             int row, int k0, int g) {
    const __bf16* Ar = A + (size_t)row * K;
    v8bf lo = *(const v8bf*)(Ar + k0 + 16 * g);
    v8bf hi = *(const v8bf*)(Ar + k0 + 16 * g + 8);
    return __builtin_shufflevector(lo, hi, 0,1,2,3,4,5,6,7,8,9,10,11,12,13,14,15);
}

// Fused-M (4 tiles) WMMA panel: one wave, one N-tile, all of M=64.
// Each B fragment is loaded once and reused by 4 WMMAs. K-unroll of 4 gives the
// scheduler a 16-WMMA / 20-load window for clause grouping and descending
// loadcnt waits while staying under 256 VGPRs (no vgpr_msb, no rotation movs).
template <int K>
__device__ __forceinline__ void wmma_panel(const __bf16* __restrict__ A,
                                           const __bf16* __restrict__ Brow,
                                           int ml, int g, v8f acc[4]) {
#pragma unroll 4
    for (int k0 = 0; k0 < K; k0 += 32) {
        v16bf b = *(const v16bf*)(Brow + k0);
#pragma unroll
        for (int m = 0; m < 4; ++m) {
            v16bf a = load_a_frag(A, K, m * 16 + ml, k0, g);
            acc[m] = __builtin_amdgcn_wmma_f32_16x16x32_bf16(false, a, false, b,
                                                             (short)0, acc[m], false, false);
        }
    }
}

// ---------------- prep kernels (run once per launch) ----------------

__global__ void prep_linW_bf16(const float* __restrict__ linW, __bf16* __restrict__ linWb) {
    int i = blockIdx.x * blockDim.x + threadIdx.x;
    if (i < VV * HH) linWb[i] = f2bf(linW[i]);
}

__global__ void prep_Wc_bf16(const float* __restrict__ Wih, const float* __restrict__ Whh,
                             __bf16* __restrict__ Wc) {
    int i = blockIdx.x * blockDim.x + threadIdx.x;
    if (i < G4H * KC) {
        int n = i >> 10;          // row in [0,2048)
        int k = i & (KC - 1);     // col in [0,1024)
        Wc[i] = (k < HH) ? f2bf(Wih[n * EE + k]) : f2bf(Whh[n * HH + (k - HH)]);
    }
}

__global__ void init_state(const float* __restrict__ features, const int* __restrict__ captions,
                           float* __restrict__ c, __bf16* __restrict__ hbf, int* __restrict__ tok) {
    int i = blockIdx.x * blockDim.x + threadIdx.x;
    if (i < BB * HH) {
        float f = features[i];
        c[i] = f;
        hbf[i] = f2bf(f);
    }
    if (i < BB) tok[i] = captions[i];   // captions shape (B,1)
}

// ---------------- per-step kernels ----------------

// Build Xcat(64 x 1024) = [ bf16(embed_W[tok[b]]) | h_bf16[b] ]
__global__ void embed_cat(const float* __restrict__ embedW, const int* __restrict__ tok,
                          const __bf16* __restrict__ hbf, __bf16* __restrict__ Xcat) {
    int i = blockIdx.x * blockDim.x + threadIdx.x;
    if (i < BB * KC) {
        int b = i >> 10;
        int k = i & (KC - 1);
        Xcat[i] = (k < EE) ? f2bf(embedW[(size_t)tok[b] * EE + k])
                           : hbf[b * HH + (k - EE)];
    }
}

// Gates(64 x 2048) = Xcat(64 x 1024) @ Wcᵀ.
__global__ __launch_bounds__(128) void gates_gemm(const __bf16* __restrict__ Xcat,
                                                  const __bf16* __restrict__ Wc,
                                                  float* __restrict__ gates) {
    int lane = threadIdx.x & 31;
    int wave = threadIdx.x >> 5;
    int n0 = (blockIdx.x * 4 + wave) * 16;
    int ml = lane & 15;
    int g  = lane >> 4;
    const __bf16* Brow = Wc + (size_t)(n0 + ml) * KC + 16 * g;

    v8f acc[4] = {v8f{}, v8f{}, v8f{}, v8f{}};
    wmma_panel<KC>(Xcat, Brow, ml, g, acc);

    int ma = g * 8;
#pragma unroll
    for (int m = 0; m < 4; ++m)
#pragma unroll
        for (int j = 0; j < 8; ++j)
            gates[(size_t)(m * 16 + ma + j) * G4H + n0 + ml] = acc[m][j];
}

// LSTM pointwise: i,f,g,o -> c_new, h_new (h stored as bf16 for next GEMMs)
__global__ void lstm_pointwise(const float* __restrict__ gates,
                               const float* __restrict__ bih, const float* __restrict__ bhh,
                               float* __restrict__ c, __bf16* __restrict__ hbf) {
    int i = blockIdx.x * blockDim.x + threadIdx.x;
    if (i >= BB * HH) return;
    int b = i >> 9;
    int j = i & (HH - 1);
    const float* gb = gates + (size_t)b * G4H;
    float ig = sigm(gb[j]            + bih[j]            + bhh[j]);
    float fg = sigm(gb[HH + j]       + bih[HH + j]       + bhh[HH + j]);
    float gg = tanhf(gb[2 * HH + j]  + bih[2 * HH + j]   + bhh[2 * HH + j]);
    float og = sigm(gb[3 * HH + j]   + bih[3 * HH + j]   + bhh[3 * HH + j]);
    float cn = fg * c[i] + ig * gg;
    c[i] = cn;
    hbf[i] = f2bf(og * tanhf(cn));
}

// Logits(64 x 32000) = h(64 x 512) @ lin_Wᵀ + lin_b  -> d_out[b][t][:]
__global__ __launch_bounds__(128) void proj_gemm(const __bf16* __restrict__ hbf,
                                                 const __bf16* __restrict__ linWb,
                                                 const float* __restrict__ linb,
                                                 float* __restrict__ out, int t) {
    int lane = threadIdx.x & 31;
    int wave = threadIdx.x >> 5;
    int n0 = (blockIdx.x * 4 + wave) * 16;
    int ml = lane & 15;
    int g  = lane >> 4;
    const __bf16* Brow = linWb + (size_t)(n0 + ml) * HH + 16 * g;

    v8f acc[4] = {v8f{}, v8f{}, v8f{}, v8f{}};
    wmma_panel<HH>(hbf, Brow, ml, g, acc);

    int ma = g * 8;
    float bias = linb[n0 + ml];
#pragma unroll
    for (int m = 0; m < 4; ++m)
#pragma unroll
        for (int j = 0; j < 8; ++j)
            out[(size_t)(m * 16 + ma + j) * (TT * VV) + (size_t)t * VV + n0 + ml] =
                acc[m][j] + bias;
}

// Per-batch-row argmax over V=32000 logits -> tok[b]  (first max, like jnp.argmax)
__global__ void row_argmax(const float* __restrict__ out, int t, int* __restrict__ tok) {
    __shared__ float sv[256];
    __shared__ int   si[256];
    int b = blockIdx.x;
    const float* row = out + (size_t)b * (TT * VV) + (size_t)t * VV;
    float best = -INFINITY;
    int   bidx = 0;
    for (int v = threadIdx.x; v < VV; v += 256) {
        float x = row[v];
        if (x > best) { best = x; bidx = v; }
    }
    sv[threadIdx.x] = best;
    si[threadIdx.x] = bidx;
    __syncthreads();
    for (int s = 128; s > 0; s >>= 1) {
        if (threadIdx.x < s) {
            float ov = sv[threadIdx.x + s];
            int   oi = si[threadIdx.x + s];
            if (ov > sv[threadIdx.x] || (ov == sv[threadIdx.x] && oi < si[threadIdx.x])) {
                sv[threadIdx.x] = ov;
                si[threadIdx.x] = oi;
            }
        }
        __syncthreads();
    }
    if (threadIdx.x == 0) tok[b] = si[0];
}

extern "C" void kernel_launch(void* const* d_in, const int* in_sizes, int n_in,
                              void* d_out, int out_size, void* d_ws, size_t ws_size,
                              hipStream_t stream) {
    (void)in_sizes; (void)n_in; (void)out_size; (void)ws_size;
    const float* features = (const float*)d_in[0];   // (B,H)
    const int*   captions = (const int*)d_in[1];     // (B,1)
    // d_in[2] = lengths (scalar, == 32, hardcoded)
    const float* embedW   = (const float*)d_in[3];   // (V,E)
    const float* Wih      = (const float*)d_in[4];   // (4H,E)
    const float* Whh      = (const float*)d_in[5];   // (4H,H)
    const float* bih      = (const float*)d_in[6];   // (4H,)
    const float* bhh      = (const float*)d_in[7];   // (4H,)
    const float* linW     = (const float*)d_in[8];   // (V,H)
    const float* linb     = (const float*)d_in[9];   // (V,)
    float* out = (float*)d_out;                      // (B,T,V)

    char* ws = (char*)d_ws;
    size_t off = 0;
    __bf16* linWb = (__bf16*)(ws + off); off += (size_t)VV * HH * 2;      // 32.768 MB
    __bf16* Wc    = (__bf16*)(ws + off); off += (size_t)G4H * KC * 2;     // 4.19 MB
    float*  gates = (float*)(ws + off);  off += (size_t)BB * G4H * 4;     // 512 KB
    float*  cst   = (float*)(ws + off);  off += (size_t)BB * HH * 4;      // 128 KB
    __bf16* hbf   = (__bf16*)(ws + off); off += (size_t)BB * HH * 2;      // 64 KB
    __bf16* Xcat  = (__bf16*)(ws + off); off += (size_t)BB * KC * 2;      // 128 KB
    int*    tok   = (int*)(ws + off);    off += 256;

    // one-time prep
    prep_linW_bf16<<<(VV * HH + 255) / 256, 256, 0, stream>>>(linW, linWb);
    prep_Wc_bf16<<<(G4H * KC + 255) / 256, 256, 0, stream>>>(Wih, Whh, Wc);
    init_state<<<(BB * HH + 255) / 256, 256, 0, stream>>>(features, captions, cst, hbf, tok);

    for (int t = 0; t < TT; ++t) {
        embed_cat<<<(BB * KC + 255) / 256, 256, 0, stream>>>(embedW, tok, hbf, Xcat);
        gates_gemm<<<dim3(G4H / 16 / 4), 128, 0, stream>>>(Xcat, Wc, gates);
        lstm_pointwise<<<(BB * HH + 255) / 256, 256, 0, stream>>>(gates, bih, bhh, cst, hbf);
        proj_gemm<<<dim3(VV / 16 / 4), 128, 0, stream>>>(hbf, linWb, linb, out, t);
        row_argmax<<<BB, 256, 0, stream>>>(out, t, tok);
    }
}
